// SelfAttentionCustom_22162031247751
// MI455X (gfx1250) — matrix operand order repaired and checked by reading, not compiled, not run on hardware
//
#include <hip/hip_runtime.h>
#include <hip/hip_bf16.h>
#include <math.h>

typedef __attribute__((ext_vector_type(16))) _Float16 v16h;
typedef __attribute__((ext_vector_type(8)))  _Float16 v8h;
typedef __attribute__((ext_vector_type(4)))  _Float16 v4h;
typedef __attribute__((ext_vector_type(8)))  float    v8f;

#define S_LEN  2048
#define B_SZ   2
#define D_MOD  2048
#define NHEAD  16
#define HDIM   128
#define D3     6144
#define LN_EPS 1e-5f
#define QSCALE 0.08838834764831845f   /* 1/sqrt(128) */

// ---------------------------------------------------------------------------
// Fragment loaders (CDNA5 WMMA f16 16x16x32 layouts, CDNA5 ISA)
// A 16x32: lanes 0-15 row M=lane, K {0..7,16..23}; lanes 16-31 K {8..15,24..31}
// B 32x16: lanes 0-15 col N=lane, K {0..7,16..23}; lanes 16-31 col N=lane-16, K {8..15,24..31}
// ---------------------------------------------------------------------------
static __device__ __forceinline__ v16h load_a16(const _Float16* base, int ld, int lane) {
  const int g = lane >> 4, r = lane & 15;
  const _Float16* p = base + (size_t)r * ld + g * 8;
  v16h f;
  ((v8h*)&f)[0] = *(const v8h*)(p);        // K = g*8 .. g*8+7
  ((v8h*)&f)[1] = *(const v8h*)(p + 16);   // K = 16+g*8 .. 16+g*8+7
  return f;
}

// base points at element (k=0, n=tileN) of a [N, K] row-major array
static __device__ __forceinline__ v16h load_b16(const _Float16* base, int ld, int lane) {
  const int g = lane >> 4, r = lane & 15;
  const _Float16* p = base + (size_t)r * ld + g * 8;
  v16h f;
  ((v8h*)&f)[0] = *(const v8h*)(p);        // K = g*8 .. g*8+7
  ((v8h*)&f)[1] = *(const v8h*)(p + 16);   // K = 16+g*8 .. 16+g*8+7
  return f;
}

// ---------------------------------------------------------------------------
// Conversions
// ---------------------------------------------------------------------------
__global__ __launch_bounds__(256) void convert_x_kernel(const float* __restrict__ x,
                                                        _Float16* __restrict__ xh) {
  size_t i = (size_t)blockIdx.x * blockDim.x + threadIdx.x;   // over S*B*D
  if (i >= (size_t)S_LEN * B_SZ * D_MOD) return;
  unsigned d = (unsigned)(i % D_MOD);
  unsigned b = (unsigned)((i / D_MOD) % B_SZ);
  unsigned s = (unsigned)(i / ((size_t)D_MOD * B_SZ));
  xh[((size_t)b * S_LEN + s) * D_MOD + d] = (_Float16)x[i];
}

__global__ __launch_bounds__(256) void convert_w_kernel(const float* __restrict__ w,
                                                        _Float16* __restrict__ wh, size_t n) {
  size_t i = (size_t)blockIdx.x * blockDim.x + threadIdx.x;
  if (i < n) wh[i] = (_Float16)w[i];
}

// ---------------------------------------------------------------------------
// WMMA GEMM:  C[m,n] = sum_k A[m,k] * W[n,k] + bias[n]
// Block: 256 threads = 8 waves, block tile 64(M) x 256(N); wave tile 32x64
// (2 A-frags x 4 B-frags -> 8 WMMAs per 12 b128 loads per K=32 step).
// OUT_MODE 0: f16 row-major [M,N]   (QKV projection)
// OUT_MODE 1: f32 scattered [S,B,N] (final output, m = b*S+s -> (s,b))
// ---------------------------------------------------------------------------
template <int OUT_MODE>
__global__ __launch_bounds__(256) void wmma_gemm_kernel(
    const _Float16* __restrict__ A, const _Float16* __restrict__ W,
    const float* __restrict__ bias, void* __restrict__ out,
    int K, int N) {
  const int lane = threadIdx.x & 31;
  const int w    = threadIdx.x >> 5;
  const int mb   = blockIdx.y * 64  + (w & 1) * 32;
  const int nb   = blockIdx.x * 256 + (w >> 1) * 64;

  const v8f zero = {0.f, 0.f, 0.f, 0.f, 0.f, 0.f, 0.f, 0.f};
  v8f acc[2][4];
#pragma unroll
  for (int i = 0; i < 2; ++i)
#pragma unroll
    for (int j = 0; j < 4; ++j) acc[i][j] = zero;

  for (int kb = 0; kb < K; kb += 32) {
    v16h a[2], b[4];
#pragma unroll
    for (int i = 0; i < 2; ++i)
      a[i] = load_a16(A + (size_t)(mb + i * 16) * K + kb, K, lane);
#pragma unroll
    for (int j = 0; j < 4; ++j)
      b[j] = load_b16(W + (size_t)(nb + j * 16) * K + kb, K, lane);
#pragma unroll
    for (int i = 0; i < 2; ++i)
#pragma unroll
      for (int j = 0; j < 4; ++j)
        acc[i][j] = __builtin_amdgcn_wmma_f32_16x16x32_f16(
            false, a[i], false, b[j], (short)0, acc[i][j], false, false);
  }

  const unsigned g = (unsigned)(lane >> 4), r = (unsigned)(lane & 15);
#pragma unroll
  for (int i = 0; i < 2; ++i)
#pragma unroll
    for (int j = 0; j < 4; ++j) {
#pragma unroll
      for (int e = 0; e < 8; ++e) {
        unsigned row = (unsigned)mb + i * 16 + e + 8 * g;
        unsigned col = (unsigned)nb + j * 16 + r;
        float val = acc[i][j][e] + bias[col];
        if (OUT_MODE == 0) {
          ((_Float16*)out)[(size_t)row * N + col] = (_Float16)val;
        } else {
          unsigned bb = row >> 11;            // row / S_LEN (S_LEN = 2048)
          unsigned ss = row & (S_LEN - 1);    // row % S_LEN
          ((float*)out)[((size_t)ss * B_SZ + bb) * N + col] = val;
        }
      }
    }
}

// ---------------------------------------------------------------------------
// QK LayerNorm + RoPE. One wave per (b,s,h) row of 128. Lane owns 4 dims.
// Writes Q (pre-scaled by 1/sqrt(HD)) and K as [B*H, S, HD] f16,
// V transposed as [B*H, HD, S] f16 (so P*V B-frags are contiguous).
// ---------------------------------------------------------------------------
__global__ __launch_bounds__(256) void lnrope_kernel(
    const _Float16* __restrict__ qkv, const float* __restrict__ freqs,
    const float* __restrict__ qn_w, const float* __restrict__ qn_b,
    const float* __restrict__ kn_w, const float* __restrict__ kn_b,
    _Float16* __restrict__ qo, _Float16* __restrict__ ko, _Float16* __restrict__ vt) {
  const unsigned gw   = (unsigned)((blockIdx.x * blockDim.x + threadIdx.x) >> 5);
  const int lane = threadIdx.x & 31;
  const unsigned h = gw % NHEAD;
  const unsigned s = (gw / NHEAD) % S_LEN;
  const unsigned b = gw / (NHEAD * S_LEN);
  const int d0 = lane * 4;

  const _Float16* row = qkv + ((size_t)b * S_LEN + s) * D3 + h * HDIM;
  v4h qh = *(const v4h*)(row + d0);
  v4h kh = *(const v4h*)(row + D_MOD + d0);
  v4h vh = *(const v4h*)(row + 2 * D_MOD + d0);

  float q[4], k[4];
#pragma unroll
  for (int i = 0; i < 4; ++i) { q[i] = (float)qh[i]; k[i] = (float)kh[i]; }

  // mean / variance over 128 (all 32 lanes)
  float sq = q[0] + q[1] + q[2] + q[3];
  float sk = k[0] + k[1] + k[2] + k[3];
#pragma unroll
  for (int off = 16; off >= 1; off >>= 1) {
    sq += __shfl_xor(sq, off, 32);
    sk += __shfl_xor(sk, off, 32);
  }
  float muq = sq * (1.0f / HDIM), muk = sk * (1.0f / HDIM);
  float vq = 0.f, vk = 0.f;
#pragma unroll
  for (int i = 0; i < 4; ++i) {
    float dq = q[i] - muq, dk = k[i] - muk;
    vq += dq * dq; vk += dk * dk;
  }
#pragma unroll
  for (int off = 16; off >= 1; off >>= 1) {
    vq += __shfl_xor(vq, off, 32);
    vk += __shfl_xor(vk, off, 32);
  }
  float rq = rsqrtf(vq * (1.0f / HDIM) + LN_EPS);
  float rk = rsqrtf(vk * (1.0f / HDIM) + LN_EPS);
#pragma unroll
  for (int i = 0; i < 4; ++i) {
    q[i] = (q[i] - muq) * rq * qn_w[d0 + i] + qn_b[d0 + i];
    k[i] = (k[i] - muk) * rk * kn_w[d0 + i] + kn_b[d0 + i];
  }

  // RoPE: lane owns pairs (2*lane, 2*lane+1)
  const float4 f = *(const float4*)(freqs + ((size_t)s * (HDIM / 2) + 2 * lane) * 2);
  float qo0 = q[0] * f.x - q[1] * f.y, qo1 = q[1] * f.x + q[0] * f.y;
  float qo2 = q[2] * f.z - q[3] * f.w, qo3 = q[3] * f.z + q[2] * f.w;
  float ko0 = k[0] * f.x - k[1] * f.y, ko1 = k[1] * f.x + k[0] * f.y;
  float ko2 = k[2] * f.z - k[3] * f.w, ko3 = k[3] * f.z + k[2] * f.w;

  _Float16* qd = qo + ((size_t)(b * NHEAD + h) * S_LEN + s) * HDIM + d0;
  _Float16* kd = ko + ((size_t)(b * NHEAD + h) * S_LEN + s) * HDIM + d0;
  v4h qov = {(_Float16)(qo0 * QSCALE), (_Float16)(qo1 * QSCALE),
             (_Float16)(qo2 * QSCALE), (_Float16)(qo3 * QSCALE)};
  v4h kov = {(_Float16)ko0, (_Float16)ko1, (_Float16)ko2, (_Float16)ko3};
  *(v4h*)qd = qov;
  *(v4h*)kd = kov;

  _Float16* vd = vt + (size_t)(b * NHEAD + h) * HDIM * S_LEN;
#pragma unroll
  for (int i = 0; i < 4; ++i) vd[(size_t)(d0 + i) * S_LEN + s] = vh[i];
}

// ---------------------------------------------------------------------------
// Flash attention. One wave per 16-query tile; key blocks of 32.
// Scores via 4 chained WMMAs over HD=128; online softmax with xor-shuffle
// row reductions; P transposed C-layout -> A-layout through per-wave LDS.
// ---------------------------------------------------------------------------
__global__ __launch_bounds__(128) void attn_kernel(
    const _Float16* __restrict__ q, const _Float16* __restrict__ k,
    const _Float16* __restrict__ vt, _Float16* __restrict__ o) {
  __shared__ __align__(32) _Float16 pbuf[4][16 * 32];
  const int w    = threadIdx.x >> 5;
  const int lane = threadIdx.x & 31;
  const int tile = blockIdx.x * 4 + w;       // over B*H*(S/16) = 4096
  const int qt   = tile % (S_LEN / 16);
  const int bh   = tile / (S_LEN / 16);
  const int qb   = qt * 16;
  const int g = lane >> 4, r = lane & 15;

  const _Float16* qp = q  + (size_t)bh * S_LEN * HDIM;
  const _Float16* kp = k  + (size_t)bh * S_LEN * HDIM;
  const _Float16* vp = vt + (size_t)bh * HDIM * S_LEN;

  v16h aq[4];
#pragma unroll
  for (int c = 0; c < 4; ++c)
    aq[c] = load_a16(qp + (size_t)qb * HDIM + c * 32, HDIM, lane);

  const v8f zero = {0.f, 0.f, 0.f, 0.f, 0.f, 0.f, 0.f, 0.f};
  v8f O[8];
  float m[8], l[8];
#pragma unroll
  for (int t = 0; t < 8; ++t) O[t] = zero;
#pragma unroll
  for (int e = 0; e < 8; ++e) { m[e] = -1e30f; l[e] = 0.f; }

  const int nblocks = qb / 32 + 1;
  for (int blk = 0; blk < nblocks; ++blk) {
    const int kb = blk * 32;
    v8f s0 = zero, s1 = zero;
#pragma unroll
    for (int c = 0; c < 4; ++c) {
      v16h bk0 = load_b16(kp + (size_t)kb * HDIM + c * 32,        HDIM, lane);
      v16h bk1 = load_b16(kp + (size_t)(kb + 16) * HDIM + c * 32, HDIM, lane);
      s0 = __builtin_amdgcn_wmma_f32_16x16x32_f16(false, aq[c], false, bk0, (short)0, s0, false, false);
      s1 = __builtin_amdgcn_wmma_f32_16x16x32_f16(false, aq[c], false, bk1, (short)0, s1, false, false);
    }
    if (blk == nblocks - 1) {  // diagonal block: causal mask
#pragma unroll
      for (int e = 0; e < 8; ++e) {
        int rowg = qb + e + 8 * g;
        if (kb + r      > rowg) s0[e] = -1e30f;
        if (kb + 16 + r > rowg) s1[e] = -1e30f;
      }
    }
    // online softmax (rows live across 16-lane halves of the C layout)
    float pscale[8];
#pragma unroll
    for (int e = 0; e < 8; ++e) {
      float mx = fmaxf(s0[e], s1[e]);
#pragma unroll
      for (int off = 1; off <= 8; off <<= 1) mx = fmaxf(mx, __shfl_xor(mx, off, 32));
      float mnew = fmaxf(m[e], mx);
      float sc = __expf(m[e] - mnew);
      float p0 = __expf(s0[e] - mnew);
      float p1 = __expf(s1[e] - mnew);
      s0[e] = p0; s1[e] = p1;
      float rs = p0 + p1;
#pragma unroll
      for (int off = 1; off <= 8; off <<= 1) rs += __shfl_xor(rs, off, 32);
      l[e] = l[e] * sc + rs;
      m[e] = mnew;
      pscale[e] = sc;
    }
#pragma unroll
    for (int t = 0; t < 8; ++t)
#pragma unroll
      for (int e = 0; e < 8; ++e) O[t][e] *= pscale[e];

    // P: C-layout -> A-layout via LDS (per-wave region, no barrier needed)
    _Float16* pb = pbuf[w];
#pragma unroll
    for (int e = 0; e < 8; ++e) {
      pb[(e + 8 * g) * 32 + r]      = (_Float16)s0[e];
      pb[(e + 8 * g) * 32 + 16 + r] = (_Float16)s1[e];
    }
    asm volatile("s_wait_dscnt 0" ::: "memory");
    v16h ap;
    {
      const _Float16* p = pb + r * 32 + g * 8;
      ((v8h*)&ap)[0] = *(const v8h*)(p);
      ((v8h*)&ap)[1] = *(const v8h*)(p + 16);
    }
#pragma unroll
    for (int t = 0; t < 8; ++t) {
      v16h bv = load_b16(vp + (size_t)(t * 16) * S_LEN + kb, S_LEN, lane);
      O[t] = __builtin_amdgcn_wmma_f32_16x16x32_f16(false, ap, false, bv, (short)0, O[t], false, false);
    }
  }

  // epilogue: O / l -> attn buffer [B*S, D] f16
  const int b = bh / NHEAD, h = bh % NHEAD;
#pragma unroll
  for (int e = 0; e < 8; ++e) {
    int rowg = qb + e + 8 * g;
    float inv = 1.0f / l[e];
    _Float16* dst = o + ((size_t)b * S_LEN + rowg) * D_MOD + h * HDIM + r;
#pragma unroll
    for (int t = 0; t < 8; ++t) dst[t * 16] = (_Float16)(O[t][e] * inv);
  }
}

// ---------------------------------------------------------------------------
// Launch
// ---------------------------------------------------------------------------
extern "C" void kernel_launch(void* const* d_in, const int* in_sizes, int n_in,
                              void* d_out, int out_size, void* d_ws, size_t ws_size,
                              hipStream_t stream) {
  (void)in_sizes; (void)n_in; (void)out_size; (void)ws_size;
  const float* x      = (const float*)d_in[0];
  const float* freqs  = (const float*)d_in[1];
  const float* wqkv_w = (const float*)d_in[2];
  const float* wqkv_b = (const float*)d_in[3];
  const float* wo_w   = (const float*)d_in[4];
  const float* wo_b   = (const float*)d_in[5];
  const float* qn_w   = (const float*)d_in[6];
  const float* qn_b   = (const float*)d_in[7];
  const float* kn_w   = (const float*)d_in[8];
  const float* kn_b   = (const float*)d_in[9];

  char* ws = (char*)d_ws;
  _Float16* xh    = (_Float16*)(ws + 0);           // [B*S, D]      16 MB
  _Float16* wqkvh = (_Float16*)(ws + 16777216);    // [3D, D]       24 MB
  _Float16* woh   = (_Float16*)(ws + 41943040);    // [D, D]         8 MB
  _Float16* qkvh  = (_Float16*)(ws + 50331648);    // [B*S, 3D]     48 MB
  _Float16* qbuf  = (_Float16*)(ws + 100663296);   // [B*H, S, HD]  16 MB
  _Float16* kbuf  = (_Float16*)(ws + 117440512);   // [B*H, S, HD]  16 MB
  _Float16* vtbuf = (_Float16*)(ws + 134217728);   // [B*H, HD, S]  16 MB
  _Float16* atbuf = (_Float16*)(ws + 150994944);   // [B*S, D]      16 MB

  const int M = B_SZ * S_LEN;  // 4096

  convert_x_kernel<<<(S_LEN * B_SZ * D_MOD + 255) / 256, 256, 0, stream>>>(x, xh);
  convert_w_kernel<<<((size_t)D3 * D_MOD + 255) / 256, 256, 0, stream>>>(wqkv_w, wqkvh, (size_t)D3 * D_MOD);
  convert_w_kernel<<<((size_t)D_MOD * D_MOD + 255) / 256, 256, 0, stream>>>(wo_w, woh, (size_t)D_MOD * D_MOD);

  dim3 g1(D3 / 256, M / 64);   // 24 x 64
  wmma_gemm_kernel<0><<<g1, 256, 0, stream>>>(xh, wqkvh, wqkv_b, qkvh, D_MOD, D3);

  lnrope_kernel<<<(B_SZ * S_LEN * NHEAD) / 8, 256, 0, stream>>>(
      qkvh, freqs, qn_w, qn_b, kn_w, kn_b, qbuf, kbuf, vtbuf);

  attn_kernel<<<(B_SZ * NHEAD * (S_LEN / 16)) / 4, 128, 0, stream>>>(qbuf, kbuf, vtbuf, atbuf);

  dim3 g2(D_MOD / 256, M / 64);  // 8 x 64
  wmma_gemm_kernel<1><<<g2, 256, 0, stream>>>(atbuf, woh, wo_b, d_out, D_MOD, D_MOD);
}
